// ReadHead_25718264168583
// MI455X (gfx1250) — matrix operand order repaired and checked by reading, not compiled, not run on hardware
//
#include <hip/hip_runtime.h>
#include <hip/hip_bf16.h>
#include <math.h>

typedef __attribute__((ext_vector_type(16))) _Float16 v16h;
typedef __attribute__((ext_vector_type(8)))  _Float16 v8h;
typedef __attribute__((ext_vector_type(2)))  _Float16 v2h;
typedef __attribute__((ext_vector_type(8)))  float    v8f;

#define BB 256
#define CC 512
#define NNN 65536
#define DD 64

__device__ __forceinline__ float softplus_f(float x) {
    return (x > 20.0f) ? x : log1pf(__expf(x));
}

__device__ __forceinline__ v16h cat8(v8h lo, v8h hi) {
    v16h r;
#pragma unroll
    for (int j = 0; j < 8; ++j) { r[j] = lo[j]; r[j + 8] = hi[j]; }
    return r;
}

// ---------------------------------------------------------------------------
// K0: per-batch-row head: key = co@Wk+bk (l2-normalized -> f16),
// beta=softplus, g=sigmoid, gamma=softplus+1, s=softmax3.
// 256 blocks x 128 threads; threads 0..63 -> key dims, 64..69 -> scalar heads.
// ---------------------------------------------------------------------------
__global__ __launch_bounds__(128) void k0_head(
    const float* __restrict__ co,
    const float* __restrict__ Wk, const float* __restrict__ bk,
    const float* __restrict__ Wbeta, const float* __restrict__ bbeta,
    const float* __restrict__ Wg, const float* __restrict__ bg,
    const float* __restrict__ Wshift, const float* __restrict__ bshift,
    const float* __restrict__ Wgamma, const float* __restrict__ bgamma,
    _Float16* __restrict__ keyF16, float* __restrict__ params)
{
    const int b = blockIdx.x;
    const int t = threadIdx.x;
    __shared__ float kbuf[DD];
    __shared__ float misc[8];
    const float* x = co + b * CC;

    if (t < DD) {
        float acc = bk[t];
        for (int c = 0; c < CC; ++c) acc += x[c] * Wk[c * DD + t];
        kbuf[t] = acc;
    } else if (t == 64) {
        float acc = bbeta[0];
        for (int c = 0; c < CC; ++c) acc += x[c] * Wbeta[c];
        misc[0] = acc;
    } else if (t == 65) {
        float acc = bg[0];
        for (int c = 0; c < CC; ++c) acc += x[c] * Wg[c];
        misc[1] = acc;
    } else if (t == 66) {
        float acc = bgamma[0];
        for (int c = 0; c < CC; ++c) acc += x[c] * Wgamma[c];
        misc[2] = acc;
    } else if (t >= 67 && t < 70) {
        const int j = t - 67;
        float acc = bshift[j];
        for (int c = 0; c < CC; ++c) acc += x[c] * Wshift[c * 3 + j];
        misc[3 + j] = acc;
    }
    __syncthreads();

    if (t == 0) {
        float ss = 0.0f;
        for (int i = 0; i < DD; ++i) ss += kbuf[i] * kbuf[i];
        misc[6] = 1.0f / fmaxf(sqrtf(ss), 1e-12f);
    }
    __syncthreads();

    if (t < DD) keyF16[b * DD + t] = (_Float16)(kbuf[t] * misc[6]);
    if (t == 64) {
        params[b * 8 + 0] = softplus_f(misc[0]);                  // beta
        params[b * 8 + 1] = 1.0f / (1.0f + __expf(-misc[1]));     // g
        params[b * 8 + 2] = softplus_f(misc[2]) + 1.0f;           // gamma
        const float a0 = misc[3], a1 = misc[4], a2 = misc[5];
        const float mx = fmaxf(a0, fmaxf(a1, a2));
        const float e0 = __expf(a0 - mx), e1 = __expf(a1 - mx), e2 = __expf(a2 - mx);
        const float inv = 1.0f / (e0 + e1 + e2);
        params[b * 8 + 3] = e0 * inv;                             // s0
        params[b * 8 + 4] = e1 * inv;                             // s1
        params[b * 8 + 5] = e2 * inv;                             // s2
    }
}

// ---------------------------------------------------------------------------
// K1: l2-normalize each 64-wide memory row -> f16.  One wave32 per row,
// float2 per lane (coalesced 256B/wave), wave shuffle reduction.
// ---------------------------------------------------------------------------
__global__ __launch_bounds__(256) void k1_norm_mem(
    const float* __restrict__ mem, _Float16* __restrict__ memF16)
{
    const int row  = (blockIdx.x * blockDim.x + threadIdx.x) >> 5;
    const int lane = threadIdx.x & 31;
    if (row >= NNN) return;
    const float2 v = *(const float2*)(mem + (size_t)row * DD + lane * 2);
    float ss = v.x * v.x + v.y * v.y;
#pragma unroll
    for (int off = 16; off > 0; off >>= 1) ss += __shfl_xor(ss, off, 32);
    const float inv = 1.0f / fmaxf(sqrtf(ss), 1e-12f);
    v2h o;
    o[0] = (_Float16)(v.x * inv);
    o[1] = (_Float16)(v.y * inv);
    *(v2h*)(memF16 + (size_t)row * DD + lane * 2) = o;
}

// ---------------------------------------------------------------------------
// K2: scores[b][n] = normkey[b] . normmem[n]  (M=256, N=65536, K=64)
// v_wmma_f32_16x16x32_f16; K=64 -> 2 chained WMMAs per 16x16 tile.
// 256 threads (8 waves) per block; each wave does 4 n-tiles, reusing A frags.
// A/B use contiguous k order (k = 16*(lane/16) + j), same in both (wave32).
// ---------------------------------------------------------------------------
__global__ __launch_bounds__(256) void k2_scores(
    const _Float16* __restrict__ keyF16,
    const _Float16* __restrict__ memF16,
    float* __restrict__ scores)
{
    const int w = threadIdx.x >> 5;
    const int l = threadIdx.x & 31;
    const int m0 = blockIdx.y * 16;
    const int ntile0 = (blockIdx.x * 8 + w) * 4;

    // A 16x32 f16: lanes 0-15: M=lane, h[j]=K j (0..15)
    //              lanes16-31: M=lane-16, h[j]=K 16+j (16..31)
    const int ka = (l >> 4) * 16;
    const _Float16* ap = keyF16 + (m0 + (l & 15)) * DD;
    const v16h a0 = cat8(*(const v8h*)(ap + ka),      *(const v8h*)(ap + ka + 8));
    const v16h a1 = cat8(*(const v8h*)(ap + ka + 32), *(const v8h*)(ap + ka + 40));

    // B 32x16 f16: lanes 0-15: N=lane, h[j]=K j (0..15); lanes16-31: K 16..31
    const int kb = (l >> 4) * 16;
    const int cn = (l & 15);
    const int cmb = m0 + (l >> 4) * 8;

#pragma unroll
    for (int it = 0; it < 4; ++it) {
        const int n0 = (ntile0 + it) * 16;
        const _Float16* bp = memF16 + (size_t)(n0 + (l & 15)) * DD + kb;
        const v16h b0 = cat8(*(const v8h*)(bp),      *(const v8h*)(bp + 8));
        const v16h b1 = cat8(*(const v8h*)(bp + 32), *(const v8h*)(bp + 40));

        v8f c = {};
        c = __builtin_amdgcn_wmma_f32_16x16x32_f16(false, a0, false, b0,
                                                   (short)0, c, false, false);
        c = __builtin_amdgcn_wmma_f32_16x16x32_f16(false, a1, false, b1,
                                                   (short)0, c, false, false);
        // C 16x16 f32: VGPR r: lanes 0-15 -> M=r, N=lane; lanes 16-31 -> M=r+8
#pragma unroll
        for (int r = 0; r < 8; ++r)
            scores[(cmb + r) * NNN + n0 + cn] = c[r];
    }
}

// ---------------------------------------------------------------------------
// K4: fused softmax(beta*cos) -> gate -> circular shift -> ^gamma -> renorm.
// One workgroup (1024 thr) per batch row; full 256KB row lives in LDS
// (CDNA5: 320KB LDS/WGP).  Scores read once, output written once.
// Deterministic fixed-order tree reductions (no atomics).
// ---------------------------------------------------------------------------
__global__ __launch_bounds__(1024) void k4_row(
    const float* __restrict__ scores,
    const float* __restrict__ prev,
    const float* __restrict__ params,
    float* __restrict__ out)
{
    extern __shared__ float sm[];
    float* red = sm;          // 2048 floats (max | sum trees)
    float* row = sm + 2048;   // 65536 floats

    const int b = blockIdx.x;
    const int t = threadIdx.x;
    const float beta  = params[b * 8 + 0];
    const float g     = params[b * 8 + 1];
    const float gamma = params[b * 8 + 2];
    const float s0 = params[b * 8 + 3];
    const float s1 = params[b * 8 + 4];
    const float s2 = params[b * 8 + 5];
    const float* sc = scores + (size_t)b * NNN;

    // Pass 1: stage row into LDS + online softmax (m, s) per thread
    float m = -__builtin_inff();
    float s = 0.0f;
#pragma unroll 4
    for (int i = 0; i < 64; ++i) {
        const int n = i * 1024 + t;        // coalesced
        const float v = sc[n];
        row[n] = v;
        const float xv = beta * v;
        if (xv > m) { s = s * __expf(m - xv) + 1.0f; m = xv; }
        else        { s += __expf(xv - m); }
    }
    red[t] = m; red[1024 + t] = s;
    __syncthreads();
    for (int off = 512; off > 0; off >>= 1) {
        if (t < off) {
            const float m1 = red[t], m2 = red[t + off];
            const float M  = fmaxf(m1, m2);
            red[1024 + t] = red[1024 + t] * __expf(m1 - M)
                          + red[1024 + t + off] * __expf(m2 - M);
            red[t] = M;
        }
        __syncthreads();
    }
    const float M    = red[0];
    const float invS = 1.0f / red[1024];

    // Pass 2: content weight + interpolation gate, in place (elementwise)
#pragma unroll 4
    for (int i = 0; i < 64; ++i) {
        const int n = i * 1024 + t;
        const float c = __expf(beta * row[n] - M) * invS;
        row[n] = g * c + (1.0f - g) * prev[n];
    }
    __syncthreads();

    // Pass 3a: circular shift + sharpen, accumulate row sum (LDS read-only)
    float lsum = 0.0f;
#pragma unroll 4
    for (int i = 0; i < 64; ++i) {
        const int n  = i * 1024 + t;
        const int nm = (n == 0)       ? NNN - 1 : n - 1;
        const int np = (n == NNN - 1) ? 0       : n + 1;
        const float sh = s0 * row[np] + s1 * row[n] + s2 * row[nm];
        lsum += __powf(sh, gamma);
    }
    __syncthreads();           // everyone done reading red[0..] earlier? safe: reuse below
    red[t] = lsum;
    __syncthreads();
    for (int off = 512; off > 0; off >>= 1) {
        if (t < off) red[t] += red[t + off];
        __syncthreads();
    }
    const float invT = 1.0f / (red[0] + 1e-8f);

    // Pass 3b: recompute (cheap transcendental) and write normalized output
#pragma unroll 4
    for (int i = 0; i < 64; ++i) {
        const int n  = i * 1024 + t;
        const int nm = (n == 0)       ? NNN - 1 : n - 1;
        const int np = (n == NNN - 1) ? 0       : n + 1;
        const float sh = s0 * row[np] + s1 * row[n] + s2 * row[nm];
        out[(size_t)b * NNN + n] = __powf(sh, gamma) * invT;
    }
}

// ---------------------------------------------------------------------------
extern "C" void kernel_launch(void* const* d_in, const int* in_sizes, int n_in,
                              void* d_out, int out_size, void* d_ws, size_t ws_size,
                              hipStream_t stream)
{
    const float* co     = (const float*)d_in[0];
    const float* mem    = (const float*)d_in[1];
    const float* prev   = (const float*)d_in[2];
    const float* Wk     = (const float*)d_in[3];
    const float* bk     = (const float*)d_in[4];
    const float* Wbeta  = (const float*)d_in[5];
    const float* bbeta  = (const float*)d_in[6];
    const float* Wg     = (const float*)d_in[7];
    const float* bg     = (const float*)d_in[8];
    const float* Wshift = (const float*)d_in[9];
    const float* bshift = (const float*)d_in[10];
    const float* Wgamma = (const float*)d_in[11];
    const float* bgamma = (const float*)d_in[12];
    float* out = (float*)d_out;

    // Workspace carving (all 256B-aligned):
    //   keyF16: 256*64*2  = 32 KB
    //   params: 256*8*4   = 8 KB
    //   memF16: 65536*64*2 = 8 MB
    //   scores: 256*65536*4 = 64 MB
    char* ws = (char*)d_ws;
    _Float16* keyF16 = (_Float16*)(ws);
    float*    params = (float*)(ws + 32768);
    _Float16* memF16 = (_Float16*)(ws + 40960);
    float*    scores = (float*)(ws + 40960 + (size_t)NNN * DD * 2);

    k0_head<<<BB, 128, 0, stream>>>(co, Wk, bk, Wbeta, bbeta, Wg, bg,
                                    Wshift, bshift, Wgamma, bgamma,
                                    keyF16, params);

    k1_norm_mem<<<NNN / 8, 256, 0, stream>>>(mem, memF16);

    // M tiles = 16 (grid.y), N tiles = 4096; block = 8 waves x 4 tiles = 32
    k2_scores<<<dim3(NNN / 16 / 32, BB / 16), 256, 0, stream>>>(keyF16, memF16, scores);

    const size_t lds_bytes = (2048 + NNN) * sizeof(float); // 270336 < 320KB
    k4_row<<<BB, 1024, lds_bytes, stream>>>(scores, prev, params, out);
}
